// Extractor_13511967113221
// MI455X (gfx1250) — matrix operand order repaired and checked by reading, not compiled, not run on hardware
//
#include <hip/hip_runtime.h>
#include <hip/hip_bf16.h>
#include <math.h>

typedef __attribute__((ext_vector_type(16))) _Float16 v16h;
typedef __attribute__((ext_vector_type(8)))  float    v8f;

__device__ __forceinline__ v8f wmma_f16(v16h a, v16h b, v8f c) {
    // D = A(16x32 f16) * B(32x16 f16) + C(16x16 f32)
    return __builtin_amdgcn_wmma_f32_16x16x32_f16(false, a, false, b, (short)0, c, false, false);
}

// A-matrix 16x32 f16 lane layout: lane m=lane%16, group g=lane/16.
// element e -> K = (e<8) ? g*8+e : 16 + g*8 + (e-8)
__device__ __forceinline__ int amapK(int e, int g) {
    return (e < 8) ? (g * 8 + e) : (16 + g * 8 + (e - 8));
}

// branch-free fast activations: one v_exp_f32 + one v_rcp_f32 each
__device__ __forceinline__ float fast_sigmoid(float v) {
    return __builtin_amdgcn_rcpf(1.f + __expf(-v));
}
__device__ __forceinline__ float fast_tanh(float v) {
    // tanh(x) = 1 - 2/(exp(2x)+1); saturates correctly at +-inf
    return 1.f - 2.f * __builtin_amdgcn_rcpf(__expf(2.f * v) + 1.f);
}
// unified: act = a*rcp(exp(s*v)+1)+b ; (s,a,b)=(2,-2,1)->tanh, (-1,1,0)->sigmoid
__device__ __forceinline__ float fast_act_sel(float v, float s, float a, float b) {
    return fmaf(__builtin_amdgcn_rcpf(__expf(s * v) + 1.f), a, b);
}

// ---------------- constants ----------------
#define BB 32
#define CR 32
#define CC 32
#define NN 128
#define TT 256
#define TL 250
#define SK 64
#define DY 20
#define ST 40
#define NCHUNK 10
#define INTERVAL 25

// ============================================================
// K1: gated dilated inception -> xc[B,CC,N,250]
// one block per (b,n); wave w handles branch w (k in {2,3,6,7})
// M=16 rows = 8 filter + 8 gate outs; K=32 (Cin) per tap; N=16 times
// ============================================================
__global__ void k_inception(const float* __restrict__ x,
    const float* fw0, const float* fw1, const float* fw2, const float* fw3,
    const float* gw0, const float* gw1, const float* gw2, const float* gw3,
    const float* fb0, const float* fb1, const float* fb2, const float* fb3,
    const float* gb0, const float* gb1, const float* gb2, const float* gb3,
    float* __restrict__ xc)
{
    __shared__ _Float16 wsh[9216];   // per branch: 16*32*k halves
    __shared__ float    bsh[64];     // [br][m]  m<8 filt bias, m>=8 gate bias
    const float* fwp[4] = {fw0, fw1, fw2, fw3};
    const float* gwp[4] = {gw0, gw1, gw2, gw3};
    const float* fbp[4] = {fb0, fb1, fb2, fb3};
    const float* gbp[4] = {gb0, gb1, gb2, gb3};
    const int ksz[4]   = {2, 3, 6, 7};
    const int broff[4] = {0, 1024, 2560, 5632};

    int tid = threadIdx.x;
    for (int idx = tid; idx < 9216; idx += 128) {
        int br = (idx < 1024) ? 0 : (idx < 2560) ? 1 : (idx < 5632) ? 2 : 3;
        int rem = idx - broff[br];
        int k = ksz[br];
        int tap = rem % k;
        int rest = rem / k;
        int c = rest & 31;
        int mm = rest >> 5;
        const float* wp = (mm < 8) ? fwp[br] : gwp[br];
        wsh[idx] = (_Float16)wp[((mm & 7) * 32 + c) * k + tap];
    }
    if (tid < 64) {
        int br = tid >> 4, mm = tid & 15;
        bsh[tid] = (mm < 8) ? fbp[br][mm] : gbp[br][mm - 8];
    }
    __syncthreads();

    int lane = tid & 31;
    int br = tid >> 5;
    int k = ksz[br], bro = broff[br];
    int b = blockIdx.x >> 7;
    int n = blockIdx.x & 127;
    int m = lane & 15, gsel = lane >> 4;
    size_t xbase = ((size_t)b * CR) * NN * TT + (size_t)n * TT;

    // per-lane-group activation coefficients (uniform within each half-wave)
    float acts = (gsel == 0) ?  2.f : -1.f;
    float acta = (gsel == 0) ? -2.f :  1.f;
    float actb = (gsel == 0) ?  1.f :  0.f;

    for (int tt = 0; tt < 16; tt++) {
        int t0 = tt * 16;
        int tcl = t0 + m; if (tcl > 249) tcl = 249;   // clamp (garbage cols masked on store)
        v8f acc = {};
        #pragma unroll 1
        for (int tap = 0; tap < k; tap++) {
            int off = 7 - k + tap;                    // crop-aligned input offset
            v16h A, Bv;
            #pragma unroll
            for (int e = 0; e < 16; e++) {
                int K = amapK(e, gsel);               // K = cin
                A[e] = wsh[bro + (m * 32 + K) * k + tap];
            }
            #pragma unroll
            for (int e = 0; e < 16; e++) {
                int c = amapK(e, gsel);               // K = cin
                Bv[e] = (_Float16)x[xbase + (size_t)c * NN * TT + tcl + off];
            }
            acc = wmma_f16(A, Bv, acc);
        }
        int t = t0 + m;
        #pragma unroll
        for (int r = 0; r < 8; r++) {
            float v = acc[r] + bsh[br * 16 + r + 8 * gsel];
            // rows0-7 filter->tanh, rows8-15 gate->sigmoid; single exp+rcp, branch-free
            float act = fast_act_sel(v, acts, acta, actb);
            float other = __shfl_xor(act, 16, 32);
            float prod = act * other;
            if (gsel == 0 && t < TL)
                xc[((size_t)(b * CC + br * 8 + r) * NN + n) * TL + t] = prod;
        }
    }
}

// ============================================================
// K2: skip conv as GEMM 64 x (B*N) x (32*250)
// block = one n-tile (16 cols), wave = one of 4 sk-tiles
// ============================================================
__global__ void k_skip(const float* __restrict__ xc, const float* __restrict__ skw,
                       const float* __restrict__ skb, float* __restrict__ skip_out)
{
    int tid = threadIdx.x, lane = tid & 31, mtile = tid >> 5;
    int b  = blockIdx.x >> 3;
    int n0 = (blockIdx.x & 7) * 16;
    int m = lane & 15, gsel = lane >> 4;
    v8f acc = {};
    #pragma unroll 1
    for (int c = 0; c < 32; c++) {
        // pull next channel's slabs into cache while this channel's WMMAs run
        if (c + 1 < 32) {
            __builtin_prefetch(&xc[((size_t)(b * CC + c + 1) * NN + n0 + m) * TL + gsel * 128], 0, 0);
            __builtin_prefetch(&skw[((size_t)(mtile * 16 + m) * 32 + c + 1) * TL + gsel * 128], 0, 0);
        }
        #pragma unroll 1
        for (int tb = 0; tb < 8; tb++) {
            int t0 = tb * 32;
            v16h A, Bv;
            #pragma unroll
            for (int e = 0; e < 16; e++) {
                int t = t0 + amapK(e, gsel);
                A[e] = (t < TL) ? (_Float16)skw[((size_t)(mtile * 16 + m) * 32 + c) * TL + t]
                                : (_Float16)0.f;
            }
            #pragma unroll
            for (int e = 0; e < 16; e++) {
                int t = t0 + amapK(e, gsel);
                Bv[e] = (t < TL) ? (_Float16)xc[((size_t)(b * CC + c) * NN + n0 + m) * TL + t]
                                 : (_Float16)0.f;
            }
            acc = wmma_f16(A, Bv, acc);
        }
    }
    int n = n0 + m;
    #pragma unroll
    for (int r = 0; r < 8; r++) {
        int sk = mtile * 16 + r + 8 * gsel;
        skip_out[((size_t)b * SK + sk) * NN + n] = acc[r] + skb[sk];
    }
}

// ============================================================
// K3: sequential GRU scan + dynamic adjacency (one block per b)
// writes row-normalized adjacency a[nchunk,B,N,N]
// all register arrays fully unrolled to avoid movrels/scratch
// ============================================================
__global__ void k_gru_adj(const float* __restrict__ xc, const float* __restrict__ st,
                          const float* __restrict__ s2dw, const float* __restrict__ s2db,
                          const float* __restrict__ wih, const float* __restrict__ whh,
                          const float* __restrict__ bih, const float* __restrict__ bhh,
                          float* __restrict__ aout)
{
    int b = blockIdx.x;
    int n = threadIdx.x;
    __shared__ float hsh[NN][DY + 1];

    float h[DY];
    #pragma unroll
    for (int d = 0; d < DY; d++) {
        float s = s2db[d];
        #pragma unroll 8
        for (int j = 0; j < ST; j++) s += st[n * ST + j] * s2dw[d * ST + j];
        h[d] = s;
    }
    #pragma unroll 1
    for (int ck = 0; ck < NCHUNK; ck++) {
        float inp[CC];
        #pragma unroll
        for (int c = 0; c < CC; c++) {
            const float* p = xc + ((size_t)(b * CC + c) * NN + n) * TL + ck * INTERVAL;
            float s = 0.f;
            #pragma unroll 5
            for (int i = 0; i < INTERVAL; i++) s += p[i];
            inp[c] = s * (1.f / INTERVAL);
        }
        float hold[DY];
        #pragma unroll
        for (int d = 0; d < DY; d++) hold[d] = h[d];
        #pragma unroll
        for (int d = 0; d < DY; d++) {
            float gir = bih[d], giz = bih[DY + d], gin = bih[2 * DY + d];
            #pragma unroll
            for (int c = 0; c < CC; c++) {
                gir += inp[c] * wih[d * CC + c];
                giz += inp[c] * wih[(DY + d) * CC + c];
                gin += inp[c] * wih[(2 * DY + d) * CC + c];
            }
            float ghr = bhh[d], ghz = bhh[DY + d], ghn = bhh[2 * DY + d];
            #pragma unroll
            for (int dd = 0; dd < DY; dd++) {
                ghr += hold[dd] * whh[d * DY + dd];
                ghz += hold[dd] * whh[(DY + d) * DY + dd];
                ghn += hold[dd] * whh[(2 * DY + d) * DY + dd];
            }
            float r = fast_sigmoid(gir + ghr);
            float z = fast_sigmoid(giz + ghz);
            float cand = fast_tanh(gin + r * ghn);
            h[d] = (1.f - z) * cand + z * hold[d];
        }
        __syncthreads();                    // prior chunk's adjacency reads done
        #pragma unroll
        for (int d = 0; d < DY; d++) hsh[n][d] = h[d];
        __syncthreads();
        float sum = 1.f;                    // +I self loop
        #pragma unroll 1
        for (int mm = 0; mm < NN; mm++) {
            float s = 0.f;
            #pragma unroll
            for (int d = 0; d < DY; d++) s += h[d] * hsh[mm][d];
            sum += fmaxf(fast_tanh(s), 0.f);
        }
        float inv = __builtin_amdgcn_rcpf(sum);
        float* ap = aout + ((size_t)(ck * BB + b) * NN + n) * NN;
        #pragma unroll 1
        for (int mm = 0; mm < NN; mm++) {
            float s = 0.f;
            #pragma unroll
            for (int d = 0; d < DY; d++) s += h[d] * hsh[mm][d];
            float rel = fmaxf(fast_tanh(s), 0.f) + ((mm == n) ? 1.f : 0.f);
            ap[mm] = rel * inv;
        }
    }
}

// ============================================================
// K4: one MixProp hop: hout = 0.05*xc + 0.95*(hin @ a)   (f16 out)
// wave = (b, chunk, ntile); GEMM 800 x 128 x 128 per (b,chunk)
// ============================================================
__global__ void k_mixprop(const float* __restrict__ xc, const float* __restrict__ hin_f,
                          const _Float16* __restrict__ hin_h, int use_half,
                          const float* __restrict__ adj, _Float16* __restrict__ hout)
{
    int tid = threadIdx.x, lane = tid & 31, widx = tid >> 5;
    int gw = blockIdx.x * 4 + widx;
    int ntile = gw % 8;
    int chunk = (gw / 8) % NCHUNK;
    int b = gw / (8 * NCHUNK);
    int m = lane & 15, gsel = lane >> 4;
    int mcol = ntile * 16 + m;

    v16h Bv[4];                                     // adjacency, K=n source node
    #pragma unroll
    for (int kc = 0; kc < 4; kc++)
        #pragma unroll
        for (int e = 0; e < 16; e++) {
            int nsrc = kc * 32 + amapK(e, gsel);
            Bv[kc][e] = (_Float16)adj[((size_t)(chunk * BB + b) * NN + nsrc) * NN + mcol];
        }

    #pragma unroll 1
    for (int rt = 0; rt < 50; rt++) {               // 800 rows = (c,t) pairs
        int row = rt * 16 + m;
        int c = row / INTERVAL, tl = row % INTERVAL;
        size_t abase = ((size_t)(b * CC + c) * NN) * TL + (size_t)(chunk * INTERVAL + tl);
        v8f acc = {};
        #pragma unroll
        for (int kc = 0; kc < 4; kc++) {
            v16h A;
            #pragma unroll
            for (int e = 0; e < 16; e++) {
                int nsrc = kc * 32 + amapK(e, gsel);
                size_t idx = abase + (size_t)nsrc * TL;
                A[e] = use_half ? hin_h[idx] : (_Float16)hin_f[idx];
            }
            acc = wmma_f16(A, Bv[kc], acc);
        }
        #pragma unroll
        for (int r = 0; r < 8; r++) {
            int rowo = rt * 16 + r + 8 * gsel;
            int co = rowo / INTERVAL, to = rowo % INTERVAL;
            size_t oidx = ((size_t)(b * CC + co) * NN + mcol) * TL + chunk * INTERVAL + to;
            hout[oidx] = (_Float16)(0.05f * xc[oidx] + 0.95f * acc[r]);
        }
    }
}

// ============================================================
// K5: 1x1 MLP over concat[x,h1,h2] (K=96) + residual -> xo (pre-LN)
// wave = (b, n, ttile, mtile); source buffer resolved per K-chunk (kc)
// ============================================================
__global__ void k_mlp(const float* __restrict__ xc, const _Float16* __restrict__ h1,
                      const _Float16* __restrict__ h2, const float* __restrict__ mlp_w,
                      const float* __restrict__ mlp_b, const float* __restrict__ x,
                      float* __restrict__ xo)
{
    __shared__ _Float16 wsh[CR * 96];
    __shared__ float    bsh[CR];
    int tid = threadIdx.x;
    for (int i = tid; i < CR * 96; i += 128) wsh[i] = (_Float16)mlp_w[i];
    if (tid < CR) bsh[tid] = mlp_b[tid];
    __syncthreads();

    int lane = tid & 31, widx = tid >> 5;
    int gw = blockIdx.x * 4 + widx;
    int mtile = gw & 1;
    int ttile = (gw >> 1) & 15;
    int n = (gw >> 5) & 127;
    int b = gw >> 12;
    int m = lane & 15, gsel = lane >> 4;
    int t0 = ttile * 16;
    int tcl = t0 + m; if (tcl > 249) tcl = 249;

    v8f acc = {};
    #pragma unroll
    for (int kc = 0; kc < 3; kc++) {
        v16h A, Bv;
        #pragma unroll
        for (int e = 0; e < 16; e++) {
            int c = kc * 32 + amapK(e, gsel);
            A[e] = wsh[(mtile * 16 + m) * 96 + c];
        }
        #pragma unroll
        for (int e = 0; e < 16; e++) {
            int cl = amapK(e, gsel);                   // local channel within chunk
            size_t idx = ((size_t)(b * CC + cl) * NN + n) * TL + tcl;
            float v;
            if (kc == 0)      v = xc[idx];             // kc selects buffer at compile time
            else if (kc == 1) v = (float)h1[idx];
            else              v = (float)h2[idx];
            Bv[e] = (_Float16)v;
        }
        acc = wmma_f16(A, Bv, acc);
    }
    int t = t0 + m;
    if (t < TL) {
        #pragma unroll
        for (int r = 0; r < 8; r++) {
            int o = mtile * 16 + r + 8 * gsel;
            float val = acc[r] + bsh[o] + x[((size_t)(b * CR + o) * NN + n) * TT + t + (TT - TL)];
            xo[((size_t)(b * CR + o) * NN + n) * TL + t] = val;
        }
    }
}

// ============================================================
// K6/K7: LayerNorm over (C,N,T) per batch
// ============================================================
#define PERB (CR * NN * TL)   // 1,024,000
__global__ void k_ln_stats(const float* __restrict__ xo, float* __restrict__ stats)
{
    int b = blockIdx.x;
    const float* p = xo + (size_t)b * PERB;
    float s1 = 0.f, s2 = 0.f;
    for (int i = threadIdx.x; i < PERB; i += 256) { float v = p[i]; s1 += v; s2 += v * v; }
    __shared__ float sh1[256], sh2[256];
    sh1[threadIdx.x] = s1; sh2[threadIdx.x] = s2;
    __syncthreads();
    for (int s = 128; s > 0; s >>= 1) {
        if (threadIdx.x < s) { sh1[threadIdx.x] += sh1[threadIdx.x + s]; sh2[threadIdx.x] += sh2[threadIdx.x + s]; }
        __syncthreads();
    }
    if (threadIdx.x == 0) {
        float mu = sh1[0] / (float)PERB;
        float var = sh2[0] / (float)PERB - mu * mu;
        stats[b * 2] = mu;
        stats[b * 2 + 1] = rsqrtf(var + 1e-5f);
    }
}

__global__ void k_ln_apply(float* __restrict__ xo, const float* __restrict__ stats,
                           const float* __restrict__ g, const float* __restrict__ be)
{
    size_t i = (size_t)blockIdx.x * blockDim.x + threadIdx.x;
    size_t stride = (size_t)gridDim.x * blockDim.x;
    for (; i < (size_t)BB * PERB; i += stride) {
        int b = (int)(i / PERB);
        int r = (int)(i % PERB);
        xo[i] = (xo[i] - stats[b * 2]) * stats[b * 2 + 1] * g[r] + be[r];
    }
}

// ============================================================
// launcher
// ============================================================
extern "C" void kernel_launch(void* const* d_in, const int* in_sizes, int n_in,
                              void* d_out, int out_size, void* d_ws, size_t ws_size,
                              hipStream_t stream)
{
    const float* x    = (const float*)d_in[0];
    const float* st   = (const float*)d_in[1];
    const float* fw[4] = {(const float*)d_in[2],  (const float*)d_in[3],  (const float*)d_in[4],  (const float*)d_in[5]};
    const float* fb[4] = {(const float*)d_in[6],  (const float*)d_in[7],  (const float*)d_in[8],  (const float*)d_in[9]};
    const float* gw[4] = {(const float*)d_in[10], (const float*)d_in[11], (const float*)d_in[12], (const float*)d_in[13]};
    const float* gb[4] = {(const float*)d_in[14], (const float*)d_in[15], (const float*)d_in[16], (const float*)d_in[17]};
    const float* skw  = (const float*)d_in[18];
    const float* skb  = (const float*)d_in[19];
    const float* s2dw = (const float*)d_in[20];
    const float* s2db = (const float*)d_in[21];
    const float* wih  = (const float*)d_in[22];
    const float* whh  = (const float*)d_in[23];
    const float* bih  = (const float*)d_in[24];
    const float* bhh  = (const float*)d_in[25];
    const float* mlpw = (const float*)d_in[26];
    const float* mlpb = (const float*)d_in[27];
    const float* lng  = (const float*)d_in[28];
    const float* lnb  = (const float*)d_in[29];

    // workspace layout (bytes)
    const size_t XCN = (size_t)BB * CC * NN * TL;            // 32,768,000 elems
    char* ws = (char*)d_ws;
    float*    xc   = (float*)ws;                              // 131.07 MB
    _Float16* h1   = (_Float16*)(ws + XCN * 4);               //  65.54 MB
    _Float16* h2   = (_Float16*)(ws + XCN * 4 + XCN * 2);     //  65.54 MB
    float*    adj  = (float*)(ws + XCN * 4 + XCN * 4);        //  20.97 MB
    float*    stats= (float*)(ws + XCN * 4 + XCN * 4 + (size_t)NCHUNK * BB * NN * NN * 4);

    float* xo       = (float*)d_out;                          // 32,768,000
    float* skip_out = (float*)d_out + XCN;                    //    262,144

    // 1) gated inception -> xc
    k_inception<<<dim3(BB * NN), dim3(128), 0, stream>>>(x,
        fw[0], fw[1], fw[2], fw[3], gw[0], gw[1], gw[2], gw[3],
        fb[0], fb[1], fb[2], fb[3], gb[0], gb[1], gb[2], gb[3], xc);

    // 2) skip conv GEMM -> skip output
    k_skip<<<dim3(BB * 8), dim3(128), 0, stream>>>(xc, skw, skb, skip_out);

    // 3) GRU scan + dynamic adjacency
    k_gru_adj<<<dim3(BB), dim3(NN), 0, stream>>>(xc, st, s2dw, s2db, wih, whh, bih, bhh, adj);

    // 4) MixProp hops
    k_mixprop<<<dim3(BB * NCHUNK * 8 / 4), dim3(128), 0, stream>>>(xc, xc, (const _Float16*)nullptr, 0, adj, h1);
    k_mixprop<<<dim3(BB * NCHUNK * 8 / 4), dim3(128), 0, stream>>>(xc, (const float*)nullptr, h1, 1, adj, h2);

    // 5) MLP + residual -> xo (pre-LN)
    k_mlp<<<dim3(BB * NN * 16 * 2 / 4), dim3(128), 0, stream>>>(xc, h1, h2, mlpw, mlpb, x, xo);

    // 6) LayerNorm
    k_ln_stats<<<dim3(BB), dim3(256), 0, stream>>>(xo, stats);
    k_ln_apply<<<dim3(16384), dim3(256), 0, stream>>>(xo, stats, lng, lnb);
}
